// TemporalResonanceLayer_45037027066029
// MI455X (gfx1250) — matrix operand order repaired and checked by reading, not compiled, not run on hardware
//
#include <hip/hip_runtime.h>
#include <hip/hip_bf16.h>
#include <math.h>

// ---------------------------------------------------------------------------
// TemporalResonanceLayer for MI455X (gfx1250, wave32, WMMA + async-to-LDS)
//   B=8, N=2048, D=1024, K=128  ->  M = 16384 rows
// ---------------------------------------------------------------------------

typedef __attribute__((ext_vector_type(16))) _Float16 v16h;
typedef __attribute__((ext_vector_type(8)))  _Float16 v8h;
typedef __attribute__((ext_vector_type(8)))  float    v8f;
typedef __attribute__((ext_vector_type(4)))  float    v4f;

#define PI_F 3.14159265358979323846f

__device__ __forceinline__ float sigm(float x) { return 1.0f / (1.0f + __expf(-x)); }
__device__ __forceinline__ float softplus_f(float x) {
    return (x > 20.0f) ? x : log1pf(__expf(x));
}

// CDNA5 async memory->LDS copy (16B per lane), tracked by ASYNCcnt.
__device__ __forceinline__ void async_copy16(const _Float16* g, _Float16* l) {
    unsigned loff = (unsigned)(size_t)l;  // flat LDS addr[31:0] = wave-relative offset
    asm volatile("global_load_async_to_lds_b128 %0, %1, off"
                 :: "v"(loff), "v"(g)
                 : "memory");
}
__device__ __forceinline__ void wait_async0() {
#if __has_builtin(__builtin_amdgcn_s_wait_asynccnt)
    __builtin_amdgcn_s_wait_asynccnt(0);
#else
    asm volatile("s_wait_asynccnt 0x0" ::: "memory");
#endif
}

// ---------------------------------------------------------------------------
// Prep kernels
// ---------------------------------------------------------------------------
__global__ void cvt_f32_to_f16(const float* __restrict__ in, _Float16* __restrict__ out, int n4) {
    int i = blockIdx.x * blockDim.x + threadIdx.x;  // one float4 per thread
    if (i < n4) {
        v4f v = *(const v4f*)(in + (size_t)i * 4);
        _Float16* o = out + (size_t)i * 4;
        o[0] = (_Float16)v.x; o[1] = (_Float16)v.y;
        o[2] = (_Float16)v.z; o[3] = (_Float16)v.w;
    }
}

// W is (Kd x N) row-major f32. out is (N x Kpad) row-major f16 (transposed, zero-padded K).
__global__ void transpose_cvt(const float* __restrict__ W, _Float16* __restrict__ out,
                              int Kd, int N, int Kpad) {
    int idx = blockIdx.x * blockDim.x + threadIdx.x;
    int total = N * Kpad;
    if (idx >= total) return;
    int n  = idx / Kpad;
    int kk = idx - n * Kpad;
    out[idx] = (kk < Kd) ? (_Float16)W[(size_t)kk * N + n] : (_Float16)0.0f;
}

// ---------------------------------------------------------------------------
// Generic WMMA GEMM:  out[M x N] = A[M x Kd](f16) * Bt[N x Kd]^T(f16) (+bias) (+resid + scale*acc)
// Block tile: 128 x 64, 8 waves (wave32), each wave 32x32 (2x2 WMMA 16x16x32 f16).
// Double-buffered LDS staging via global_load_async_to_lds_b128 (ASYNCcnt).
// ---------------------------------------------------------------------------
template <bool HAS_BIAS, bool HAS_RES>
__global__ __launch_bounds__(256)
void gemm_wmma(const _Float16* __restrict__ A,
               const _Float16* __restrict__ Bt,
               const float* __restrict__ bias,
               const float* __restrict__ resid,
               const float* __restrict__ scale_ptr,
               float* __restrict__ out,
               int M_, int N_, int Kd) {
    (void)M_;
    __shared__ __align__(32) _Float16 sA[2][128][48];  // 96B row stride (32B-aligned)
    __shared__ __align__(32) _Float16 sB[2][64][48];

    const int tid  = threadIdx.x;
    const int lane = tid & 31;
    const int wid  = tid >> 5;           // 0..7
    const int ml   = lane & 15;          // column / M-row within fragment
    const int kh   = lane >> 4;          // K-half selector

    const int m0 = blockIdx.y * 128;
    const int n0 = blockIdx.x * 64;

    const int wave_m = wid & 3;          // 0..3  -> 32-row group
    const int wave_n = wid >> 2;         // 0..1  -> 32-col group
    const int r_base = wave_m * 32;
    const int c_base = wave_n * 32;

    v8f acc[2][2] = {};

    // async staging coords (16B chunks, one b128 per lane per chunk)
    const int a_row0 = tid >> 2;             // chunks 0..255 : rows 0..63
    const int a_row1 = (tid + 256) >> 2;     // chunks 256..511 : rows 64..127
    const int a_col  = (tid & 3) * 8;
    const int b_row  = tid >> 2;             // 0..63
    const int b_col  = (tid & 3) * 8;

    auto stage = [&](int buf, int k0) {
        async_copy16(A + (size_t)(m0 + a_row0) * Kd + k0 + a_col, &sA[buf][a_row0][a_col]);
        async_copy16(A + (size_t)(m0 + a_row1) * Kd + k0 + a_col, &sA[buf][a_row1][a_col]);
        async_copy16(Bt + (size_t)(n0 + b_row) * Kd + k0 + b_col, &sB[buf][b_row][b_col]);
    };

    const int nk = Kd >> 5;
    stage(0, 0);
    wait_async0();
    __syncthreads();

    for (int it = 0; it < nk; ++it) {
        const int cur = it & 1;
        if (it + 1 < nk) stage(cur ^ 1, (it + 1) << 5);  // prefetch next tile into other buffer

        // A fragments: lane holds row (r_base + i*16 + ml), K-halves split by kh
        v16h af[2];
#pragma unroll
        for (int i = 0; i < 2; ++i) {
            int r = r_base + i * 16 + ml;
            af[i] = *(const v16h*)&sA[cur][r][kh * 16];
        }
        // B fragments: lane holds column (c_base + j*16 + ml), contiguous 16 K values
        v16h bf[2];
#pragma unroll
        for (int j = 0; j < 2; ++j) {
            int c = c_base + j * 16 + ml;
            bf[j] = *(const v16h*)&sB[cur][c][kh * 16];
        }
#pragma unroll
        for (int i = 0; i < 2; ++i)
#pragma unroll
            for (int j = 0; j < 2; ++j)
                acc[i][j] = __builtin_amdgcn_wmma_f32_16x16x32_f16(
                    false, af[i], false, bf[j], (short)0, acc[i][j], false, false);

        if (it + 1 < nk) wait_async0();  // next tile resident before the barrier
        __syncthreads();
    }

    float sc = 1.0f;
    if (HAS_RES) sc = *scale_ptr;

#pragma unroll
    for (int i = 0; i < 2; ++i) {
#pragma unroll
        for (int j = 0; j < 2; ++j) {
            int col = n0 + c_base + j * 16 + ml;
            float bv = HAS_BIAS ? bias[col] : 0.0f;
#pragma unroll
            for (int v = 0; v < 8; ++v) {
                int row = m0 + r_base + i * 16 + kh * 8 + v;
                float val = acc[i][j][v] + bv;
                if (HAS_RES) val = resid[(size_t)row * N_ + col] + sc * val;
                out[(size_t)row * N_ + col] = val;
            }
        }
    }
}

// ---------------------------------------------------------------------------
// Scan pass 1: per-chunk local recurrence (chunk = 128 positions)
//   r[t] = alpha[t]*r[t-1] + drive[t], carry-free; store r_local and prefix
//   alpha product. Threads = k (coalesced over the k-major projection).
// ---------------------------------------------------------------------------
__global__ __launch_bounds__(128)
void scan_local(const float* __restrict__ proj, const float* __restrict__ omega_base,
                float* __restrict__ rloc_re, float* __restrict__ rloc_im,
                float* __restrict__ aprod) {
    const int k = threadIdx.x;       // 0..127
    const int chunk = blockIdx.x;    // 0..15
    const int b = blockIdx.y;        // 0..7
    const float om_b = omega_base[k];

    float rre = 0.0f, rim = 0.0f, ap = 1.0f;
    for (int nl = 0; nl < 128; ++nl) {
        int n = chunk * 128 + nl;
        size_t rowp = ((size_t)b * 2048 + n) * 768;
        __builtin_prefetch(proj + rowp + 4 * 768 + k, 0, 0);  // global_prefetch next rows
        float A_r = proj[rowp + k];
        float Om  = proj[rowp + 128 + k];
        float Ph  = proj[rowp + 256 + k];
        float Ga  = proj[rowp + 384 + k];

        float Aamp  = fminf(softplus_f(A_r), 3.0f);
        float alpha = sigm(Ga);
        float omega = fminf(fmaxf(sigm(Om) * PI_F + om_b, 1e-4f), PI_F - 1e-4f);
        float phi   = tanhf(Ph) * PI_F;
        float pos   = log1pf((float)n);
        float sa, ca;
        __sincosf(omega * pos + phi, &sa, &ca);
        float drive = (1.0f - alpha) * Aamp;

        rre = alpha * rre + drive * ca;
        rim = alpha * rim + drive * sa;
        ap *= alpha;

        size_t idx = ((size_t)b * 2048 + n) * 128 + k;
        rloc_re[idx] = rre;
        rloc_im[idx] = rim;
        aprod[idx]   = ap;
    }
}

// ---------------------------------------------------------------------------
// Scan pass 2: serial scan of 16 chunk summaries per channel (1024 channels)
// ---------------------------------------------------------------------------
__global__ void scan_carry(const float* __restrict__ rloc_re, const float* __restrict__ rloc_im,
                           const float* __restrict__ aprod,
                           float* __restrict__ car_re, float* __restrict__ car_im) {
    int tid = blockIdx.x * blockDim.x + threadIdx.x;
    if (tid >= 8 * 128) return;
    int b = tid >> 7, k = tid & 127;
    float cre = 0.0f, cim = 0.0f;
    for (int c = 0; c < 16; ++c) {
        int ci = (b * 16 + c) * 128 + k;
        car_re[ci] = cre;
        car_im[ci] = cim;
        size_t idx = ((size_t)b * 2048 + c * 128 + 127) * 128 + k;
        float Ac = aprod[idx];
        cre = Ac * cre + rloc_re[idx];
        cim = Ac * cim + rloc_im[idx];
    }
}

// ---------------------------------------------------------------------------
// Scan pass 3: apply carries + full elementwise epilogue -> feats (f16, 512 cols)
//   feats = [g*rho_re(128) | g*rho_im(128) | cross_re(127) | cross_im(127) | pad(2)]
// ---------------------------------------------------------------------------
__global__ __launch_bounds__(128)
void scan_apply_feats(const float* __restrict__ proj, const float* __restrict__ pq,
                      const float* __restrict__ omega_base,
                      const float* __restrict__ rloc_re, const float* __restrict__ rloc_im,
                      const float* __restrict__ aprod,
                      const float* __restrict__ car_re, const float* __restrict__ car_im,
                      const float* __restrict__ lambda_ptr,
                      _Float16* __restrict__ feats) {
    __shared__ float s_re[128];
    __shared__ float s_im[128];
    const int k = threadIdx.x;
    const int chunk = blockIdx.x;
    const int b = blockIdx.y;
    const float lam = *lambda_ptr;
    const float om_b = omega_base[k];

    int ci = (b * 16 + chunk) * 128 + k;
    const float cre = car_re[ci];
    const float cim = car_im[ci];

    for (int nl = 0; nl < 128; ++nl) {
        int n = chunk * 128 + nl;
        size_t rowp = ((size_t)b * 2048 + n) * 768;
        __builtin_prefetch(proj + rowp + 4 * 768 + k, 0, 0);
        float Om = proj[rowp + 128 + k];
        float Ph = proj[rowp + 256 + k];
        float Go = proj[rowp + 512 + k];
        float Be = proj[rowp + 640 + k];

        float omega = fminf(fmaxf(sigm(Om) * PI_F + om_b, 1e-4f), PI_F - 1e-4f);
        float phi   = tanhf(Ph) * PI_F;
        float pos   = log1pf((float)n);
        float sa, ca;
        __sincosf(omega * pos + phi, &sa, &ca);

        size_t idx = ((size_t)b * 2048 + n) * 128 + k;
        float ap = aprod[idx];
        float rre = cre * ap + rloc_re[idx];
        float rim = cim * ap + rloc_im[idx];

        float readout = rre * ca + rim * sa;
        float beta = sigm(Be);
        rre -= beta * readout * ca;
        rim -= beta * readout * sa;

        float modulus = sqrtf(rre * rre + rim * rim + 1e-8f);
        float inv = 1.0f / fmaxf(modulus, 1.0f);
        rre *= inv;
        rim *= inv;

        float rho_re = rre * ca + rim * sa;
        float rho_im = -rre * sa + rim * ca;
        float rho_norm = sqrtf(rho_re * rho_re + rho_im * rho_im + 1e-6f);

        float pqv = pq[idx];
        float sp, cp;
        __sincosf(pqv, &sp, &cp);
        float align = (rho_re * cp + rho_im * sp) / rho_norm;
        float gate = sigm(lam * align);
        rho_re *= gate;
        rho_im *= gate;
        float g_out = sigm(Go);

        s_re[k] = rho_re;
        s_im[k] = rho_im;
        __syncthreads();
        float nre = (k < 127) ? s_re[k + 1] : 0.0f;
        float nim = (k < 127) ? s_im[k + 1] : 0.0f;
        float cross_re = rho_re * nre + rho_im * nim;
        float cross_im = rho_im * nre - rho_re * nim;

        _Float16* f = feats + ((size_t)b * 2048 + n) * 512;
        f[k]       = (_Float16)(g_out * rho_re);
        f[128 + k] = (_Float16)(g_out * rho_im);
        if (k < 127) {
            f[256 + k] = (_Float16)cross_re;
            f[383 + k] = (_Float16)cross_im;
        }
        if (k == 127) {
            f[510] = (_Float16)0.0f;
            f[511] = (_Float16)0.0f;
        }
        __syncthreads();
    }
}

// ---------------------------------------------------------------------------
// Host-side launch
// ---------------------------------------------------------------------------
extern "C" void kernel_launch(void* const* d_in, const int* in_sizes, int n_in,
                              void* d_out, int out_size, void* d_ws, size_t ws_size,
                              hipStream_t stream) {
    (void)in_sizes; (void)n_in; (void)out_size; (void)ws_size;

    const float* x          = (const float*)d_in[0];  // (8,2048,1024)
    const float* W_proj     = (const float*)d_in[1];  // (1024,768)
    const float* b_proj     = (const float*)d_in[2];  // (768)
    const float* omega_base = (const float*)d_in[3];  // (128)
    const float* W_phase    = (const float*)d_in[4];  // (1024,128)
    const float* b_phase    = (const float*)d_in[5];  // (128)
    const float* W_res      = (const float*)d_in[6];  // (510,1024)
    const float* lambda_pcg = (const float*)d_in[7];  // scalar
    const float* res_scale  = (const float*)d_in[8];  // scalar
    float* out = (float*)d_out;

    const int M = 8 * 2048;       // 16384
    const int D = 1024;
    const int N6 = 768;           // 6*K
    const int KK = 128;
    const int KRp = 512;          // padded feature K (510 -> 512)

    // workspace carve-out
    char* base = (char*)d_ws;
    size_t off = 0;
    auto carve = [&](size_t bytes) -> char* {
        char* p = base + off;
        off = (off + bytes + 255) & ~(size_t)255;
        return p;
    };
    _Float16* x16    = (_Float16*)carve((size_t)M * D * 2);
    _Float16* wp_t   = (_Float16*)carve((size_t)N6 * D * 2);
    _Float16* wph_t  = (_Float16*)carve((size_t)KK * D * 2);
    _Float16* wres_t = (_Float16*)carve((size_t)D * KRp * 2);
    float* proj      = (float*)carve((size_t)M * N6 * 4);
    float* pqbuf     = (float*)carve((size_t)M * KK * 4);
    float* rloc_re   = (float*)carve((size_t)M * KK * 4);
    float* rloc_im   = (float*)carve((size_t)M * KK * 4);
    float* aprod     = (float*)carve((size_t)M * KK * 4);
    float* car_re    = (float*)carve((size_t)8 * 16 * 128 * 4);
    float* car_im    = (float*)carve((size_t)8 * 16 * 128 * 4);
    _Float16* feats  = (_Float16*)carve((size_t)M * KRp * 2);

    // 1) precision conversion + weight transposes
    cvt_f32_to_f16<<<(M * D / 4 + 255) / 256, 256, 0, stream>>>(x, x16, M * D / 4);
    transpose_cvt<<<(N6 * D + 255) / 256, 256, 0, stream>>>(W_proj, wp_t, D, N6, D);
    transpose_cvt<<<(KK * D + 255) / 256, 256, 0, stream>>>(W_phase, wph_t, D, KK, D);
    transpose_cvt<<<(D * KRp + 255) / 256, 256, 0, stream>>>(W_res, wres_t, 510, D, KRp);

    // 2) projection GEMMs (WMMA f16 -> f32)
    gemm_wmma<true, false><<<dim3(N6 / 64, M / 128), 256, 0, stream>>>(
        x16, wp_t, b_proj, nullptr, nullptr, proj, M, N6, D);
    gemm_wmma<true, false><<<dim3(KK / 64, M / 128), 256, 0, stream>>>(
        x16, wph_t, b_phase, nullptr, nullptr, pqbuf, M, KK, D);

    // 3) decoupled linear-recurrence scan over n (16 chunks of 128)
    scan_local<<<dim3(16, 8), 128, 0, stream>>>(proj, omega_base, rloc_re, rloc_im, aprod);
    scan_carry<<<4, 256, 0, stream>>>(rloc_re, rloc_im, aprod, car_re, car_im);
    scan_apply_feats<<<dim3(16, 8), 128, 0, stream>>>(
        proj, pqbuf, omega_base, rloc_re, rloc_im, aprod, car_re, car_im, lambda_pcg, feats);

    // 4) final GEMM + residual:  out = x + res_scale * (feats @ W_res)
    gemm_wmma<false, true><<<dim3(D / 64, M / 128), 256, 0, stream>>>(
        feats, wres_t, nullptr, x, res_scale, out, M, D, KRp);
}
